// SNNConv2d_1692217115108
// MI455X (gfx1250) — matrix operand order repaired and checked by reading, not compiled, not run on hardware
//
#include <hip/hip_runtime.h>

typedef __attribute__((ext_vector_type(16))) _Float16 v16h;
typedef __attribute__((ext_vector_type(8)))  _Float16 v8h;
typedef __attribute__((ext_vector_type(2)))  __fp16   v2hf;  // cvt_pkrtz result type
typedef __attribute__((ext_vector_type(4)))  __fp16   v4hf;
typedef __attribute__((ext_vector_type(8)))  float    v8f;
typedef __attribute__((ext_vector_type(4)))  float    v4f;

#define B_N    8
#define C_IN   16
#define H_IN   64
#define W_IN   64
#define T_N    32
#define C_OUT  32
#define KDIM   (C_IN * 9)      // 144 = Cin * 3 * 3
#define KPAD   168             // padded K stride in halves (mult of 8 -> 16B rows)
#define KSTEPS 5               // ceil(144 / 32)
#define PIX_PER_BLK 8
#define CSTRIDE 36             // f32 stride for conv[co][t] buffer (144B rows)
#define DECAY  0.9f
#define THRESH 0.25f

__global__ __launch_bounds__(32)
void snn_conv2d_wmma(const float* __restrict__ x,
                     const float* __restrict__ w,
                     float* __restrict__ out)
{
    // s_bk: first holds weights as f16 [m][KPAD], then reused per pixel as the
    //       transposed im2col patch f16 [t][KPAD].
    __shared__ __align__(16) _Float16 s_bk[32 * KPAD];         // 10.5 KB
    __shared__ __align__(16) float    s_wstage[C_OUT * KDIM];  // 18 KB (f32 weights)
    __shared__ __align__(16) float    s_conv[C_OUT * CSTRIDE]; // 4.5 KB

    const int lane = threadIdx.x;      // 0..31, one wave per block
    const int half = lane >> 4;        // 0 | 1
    const int l15  = lane & 15;

    const int wblk = blockIdx.x * PIX_PER_BLK;
    const int h    = blockIdx.y;
    const int b    = blockIdx.z;

    // ---- async DMA the f32 weights into LDS (CDNA5 async-to-LDS engine) ----
    {
        unsigned ldsw = (unsigned)(size_t)(void*)s_wstage;
        unsigned voff = (unsigned)lane * 16u;
        for (int it = 0; it < (C_OUT * KDIM * 4) / 512; ++it) {  // 36 x 512B
            asm volatile("global_load_async_to_lds_b128 %0, %1, %2"
                         :: "v"(ldsw + voff), "v"(voff), "s"(w) : "memory");
            voff += 512u;
        }
    }

    // ---- zero the K-pad region [144,168) of every row (covers A and B pads) ----
    {
        uint4 z; z.x = z.y = z.z = z.w = 0u;
        *(uint4*)(&s_bk[lane * KPAD + 144]) = z;
        *(uint4*)(&s_bk[lane * KPAD + 152]) = z;
        *(uint4*)(&s_bk[lane * KPAD + 160]) = z;
    }

    asm volatile("s_wait_asynccnt 0x0" ::: "memory");
    __syncthreads();

    // ---- convert weights f32 -> packed f16 [m][KPAD] ----
    for (int p = lane; p < (C_OUT * KDIM) / 2; p += 32) {
        int e = 2 * p;
        int m = e / KDIM;
        int k = e - m * KDIM;                 // even; pairs never straddle m (KDIM even)
        v2hf pk = __builtin_amdgcn_cvt_pkrtz(s_wstage[e], s_wstage[e + 1]);
        *(v2hf*)&s_bk[m * KPAD + k] = pk;
    }
    __syncthreads();

    // ---- build A fragments in registers (ISA 16-bit A layout) ----
    // lanes 0-15: M=0..15, halves = K 0-7 then K 16-23 (per 32-K step)
    // lanes16-31: M=0..15, halves = K 8-15 then K 24-31
    v16h afrag[2][KSTEPS];
#pragma unroll
    for (int mt = 0; mt < 2; ++mt) {
#pragma unroll
        for (int ks = 0; ks < KSTEPS; ++ks) {
            const _Float16* p = &s_bk[(mt * 16 + l15) * KPAD + ks * 32 + half * 8];
            v8h lo = *(const v8h*)p;          // K = base + 0..7
            v8h hi = *(const v8h*)(p + 16);   // K = base + 16..23
            v16h a;
#pragma unroll
            for (int i = 0; i < 8; ++i) { a[i] = lo[i]; a[i + 8] = hi[i]; }
            afrag[mt][ks] = a;
        }
    }
    __syncthreads();

    const int t4 = (lane & 7) * 4;    // t offset for vectorized x loads / out stores
    const int g4 = (lane >> 3) * 4;   // 4-row group owned by this lane

    for (int p = 0; p < PIX_PER_BLK; ++p) {
        const int wx0 = wblk + p;

        if (p + 1 < PIX_PER_BLK) {   // warm next pixel's center column
            __builtin_prefetch(x + ((((size_t)b * C_IN) * H_IN + h) * W_IN + (wx0 + 1)) * T_N, 0, 1);
        }

        auto load_row = [&](int row) -> v4f {
            int cin = row / 9;
            int rem = row - cin * 9;
            int kh  = rem / 3;
            int kw  = rem - kh * 3;
            int hy  = h   + kh - 1;
            int wxx = wx0 + kw - 1;
            v4f v = {0.f, 0.f, 0.f, 0.f};
            if ((unsigned)hy < (unsigned)H_IN && (unsigned)wxx < (unsigned)W_IN)
                v = *(const v4f*)(x + ((((size_t)b * C_IN + cin) * H_IN + hy) * W_IN + wxx) * T_N + t4);
            return v;
        };

        // ---- stage im2col patch transposed: s_bk[t][k] (f16) ----
        // each lane: 4 adjacent k-rows at its 4 t's -> pack pairs, ds_store_b64
        for (int rr = 0; rr < KDIM; rr += 16) {
            int r0 = rr + g4;
            v4f va = load_row(r0 + 0);
            v4f vb = load_row(r0 + 1);
            v4f vc = load_row(r0 + 2);
            v4f vd = load_row(r0 + 3);
#pragma unroll
            for (int j = 0; j < 4; ++j) {
                v2hf p0 = __builtin_amdgcn_cvt_pkrtz(va[j], vb[j]);  // k = r0, r0+1
                v2hf p1 = __builtin_amdgcn_cvt_pkrtz(vc[j], vd[j]);  // k = r0+2, r0+3
                v4hf q; q[0] = p0[0]; q[1] = p0[1]; q[2] = p1[0]; q[3] = p1[1];
                *(v4hf*)&s_bk[(t4 + j) * KPAD + r0] = q;             // 8B store
            }
        }
        __syncthreads();

        // ---- implicit GEMM: conv[32 cout][32 t] = W[32x144] * patch[144x32] ----
        v8f acc[2][2];
#pragma unroll
        for (int mt = 0; mt < 2; ++mt)
#pragma unroll
            for (int nt = 0; nt < 2; ++nt)
                acc[mt][nt] = (v8f){};

#pragma unroll
        for (int ks = 0; ks < KSTEPS; ++ks) {
            // B fragments (ISA B layout): lanes 0-15 hold K 0-7,16-23, lanes 16-31 hold K 8-15,24-31
            v16h bfr[2];
#pragma unroll
            for (int nt = 0; nt < 2; ++nt) {
                const int tcol = l15 + nt * 16;
                const _Float16* bp = &s_bk[tcol * KPAD + ks * 32 + half * 8];
                v8h blo = *(const v8h*)bp;
                v8h bhi = *(const v8h*)(bp + 16);
                v16h bv;
#pragma unroll
                for (int i = 0; i < 8; ++i) { bv[i] = blo[i]; bv[i + 8] = bhi[i]; }
                bfr[nt] = bv;
            }
#pragma unroll
            for (int mt = 0; mt < 2; ++mt)
#pragma unroll
                for (int nt = 0; nt < 2; ++nt)
                    acc[mt][nt] = __builtin_amdgcn_wmma_f32_16x16x32_f16(
                        false, afrag[mt][ks], false, bfr[nt],
                        (short)0, acc[mt][nt], false, false);
        }

        // ---- scatter D tiles to LDS conv[co][t] (f32) ----
        // D layout: lane = N(t) within tile, vgpr r -> M = r (lanes 0-15) / r+8 (16-31)
#pragma unroll
        for (int mt = 0; mt < 2; ++mt)
#pragma unroll
            for (int nt = 0; nt < 2; ++nt)
#pragma unroll
                for (int r = 0; r < 8; ++r) {
                    int co = mt * 16 + half * 8 + r;
                    int t  = nt * 16 + l15;
                    s_conv[co * CSTRIDE + t] = acc[mt][nt][r];
                }
        __syncthreads();

        // ---- sequential membrane scan over t: one lane per cout ----
        {
            float mem = 0.f;
            float* rowp = &s_conv[lane * CSTRIDE];
#pragma unroll
            for (int t = 0; t < T_N; ++t) {
                float m2    = (mem + rowp[t]) * DECAY;
                float spike = (m2 >= THRESH) ? 1.f : 0.f;
                rowp[t] = spike;
                mem = (spike > 0.f) ? 0.f : m2;
            }
        }
        __syncthreads();

        // ---- coalesced spike store: out[b][co][h][wx0][t], T contiguous ----
#pragma unroll
        for (int c4 = 0; c4 < C_OUT; c4 += 4) {
            int co = c4 + (lane >> 3);
            v4f v = *(const v4f*)&s_conv[co * CSTRIDE + t4];
            float* gp = out + ((((size_t)b * C_OUT + co) * H_IN + h) * W_IN + wx0) * T_N + t4;
            *(v4f*)gp = v;
        }
        __syncthreads();
    }
}

extern "C" void kernel_launch(void* const* d_in, const int* in_sizes, int n_in,
                              void* d_out, int out_size, void* d_ws, size_t ws_size,
                              hipStream_t stream) {
    const float* x = (const float*)d_in[0];   // (8,16,64,64,32) f32 spikes
    const float* w = (const float*)d_in[1];   // (32,16,3,3) f32
    float* out = (float*)d_out;               // (8,32,64,64,32) f32 spikes

    dim3 grid(W_IN / PIX_PER_BLK, H_IN, B_N); // (8, 64, 8) = 4096 blocks
    snn_conv2d_wmma<<<grid, 32, 0, stream>>>(x, w, out);
}
